// KalmanPredictorMambaRecursiveV1a_80977313399543
// MI455X (gfx1250) — matrix operand order repaired and checked by reading, not compiled, not run on hardware
//
#include <hip/hip_runtime.h>
#include <hip/hip_bf16.h>
#include <math.h>

// ---------------------------------------------------------------------------
// SS2D (VMamba) pipeline for gfx1250 / MI455X.
//  * GEMMs: v_wmma_f32_16x16x32_f16, one wave32 per 16x16 tile, guard-free
//    (shapes padded in workspace), float4 A loads, and the whole KxN=Kx16
//    B panel staged into LDS by ONE Tensor-Data-Mover tensor_load_to_lds
//    (TENSORcnt-tracked), so the K loop has zero staging/barriers.
//  * Selective scan: 16 SSM states in registers, exp() on the TRANS pipe.
//  * Workspace: ~43.3 MB of d_ws (floats).
// ---------------------------------------------------------------------------

typedef _Float16     v16h __attribute__((ext_vector_type(16)));
typedef float        v8f  __attribute__((ext_vector_type(8)));
typedef unsigned int v4u  __attribute__((ext_vector_type(4)));
typedef int          v8i  __attribute__((ext_vector_type(8)));
typedef int          v4i  __attribute__((ext_vector_type(4)));

#define C_    64
#define D_    128        // EXPAND * C
#define H_    64
#define W_    64
#define L_    4096       // H*W
#define R_    4          // dt rank = C/16
#define NS_   16         // D_STATE
#define KD_   4          // scan directions
#define XPC_  36         // r + 2n rows of x_dbl (true)
#define XPCP_ 48         // padded to multiple of 16
#define KDT_  32         // dt GEMM K padded to 32
#define KMAX_ 128        // max GEMM K (LDS B panel rows)

// --------------------- TDM: load K x 16 f32 tile into LDS ------------------
// D# per CDNA5 ISA:
//  group0: count=1 | lds_addr | global_addr[56:0] | type=2
//  group1: data_size=2 (4B) | tensor_dim0=row_stride | tensor_dim1=rows
//          | tile_dim0=16 | tile_dim1=rows | tensor_dim0_stride=row_stride
// 2-D tensor -> groups 2/3 zero.  Tile rows packed contiguously into LDS.
// This toolchain's builtin is the 6-arg form:
//   (uint32x4 g0, int32x8 g1, int32x4, int32x4, int32x8, i32 cpol)
__device__ __forceinline__ void tdm_load_b_panel(const float* gsrc,
                                                 unsigned lds_byte_off,
                                                 int rows, int row_stride) {
  unsigned long long ga = (unsigned long long)gsrc;
  v4u g0;
  g0[0] = 1u;                                        // count=1 (valid user D#)
  g0[1] = lds_byte_off;                              // lds_addr (bytes)
  g0[2] = (unsigned)(ga & 0xFFFFFFFFu);              // global_addr[31:0]
  g0[3] = (unsigned)((ga >> 32) & 0x1FFFFFFu)        // global_addr[56:32]
        | (2u << 30);                                // type = 2 ("image")
  v8i g1;
  g1[0] = (int)(2u << 16);                           // data_size=4B; mask=0
  g1[1] = (int)((unsigned)row_stride << 16);         // tensor_dim0 lo16 @63:48
  g1[2] = (int)(((unsigned)row_stride >> 16) & 0xFFFFu)  // tensor_dim0 hi16
        | (int)((unsigned)rows << 16);               // tensor_dim1 lo16 @111:80
  g1[3] = (int)(((unsigned)rows >> 16) & 0xFFFFu)    // tensor_dim1 hi16
        | (int)(16u << 16);                          // tile_dim0=16 @127:112
  g1[4] = rows;                                      // tile_dim1; tile_dim2=0
  g1[5] = row_stride;                                // tensor_dim0_stride lo32
  g1[6] = 0;                                         // stride hi / dim1_stride
  g1[7] = 0;
  v4i z4 = {0, 0, 0, 0};
  v8i z8 = {0, 0, 0, 0, 0, 0, 0, 0};
  __builtin_amdgcn_tensor_load_to_lds(g0, g1, z4, z4, z8, 0);
}

// ------------------------------- WMMA GEMM ---------------------------------
// out[M x N] = A[M x K] (row-major) * B[K x N] (row-major) + bias[m]
// Requires: M % 16 == 0, N % 16 == 0, K % 32 == 0, K <= 128, bias non-null.
// ACT: 0 = none, 1 = softplus.  Grid: (N/16, M/16), block: 32 (one wave32).
template <int ACT>
__launch_bounds__(32)
__global__ void k_wmma_gemm(const float* __restrict__ A,
                            const float* __restrict__ B,
                            const float* __restrict__ bias,
                            float* __restrict__ Out,
                            int M, int N, int K) {
  __shared__ float lb[KMAX_ * 16];        // B panel, K rows x 16 cols (<=8KB)
  const int lane = threadIdx.x;
  const int lhi  = lane >> 4;             // half-wave select
  const int lm   = lane & 15;
  const int m0   = blockIdx.y * 16;
  const int n0   = blockIdx.x * 16;
  const float* Arow = A + (size_t)(m0 + lm) * K;

  // One TDM DMA stages the whole K x 16 B panel into LDS (static LDS
  // allocation starts at offset 0; lb is the only shared object).
  tdm_load_b_panel(B + n0, 0u, K, N);

  // 8 contiguous bias values for this lane's output rows (2x b128, hoisted)
  float4 bs0 = *(const float4*)(bias + m0 + 8 * lhi);
  float4 bs1 = *(const float4*)(bias + m0 + 8 * lhi + 4);
  float bv[8] = {bs0.x, bs0.y, bs0.z, bs0.w, bs1.x, bs1.y, bs1.z, bs1.w};

  __builtin_amdgcn_s_wait_tensorcnt(0);   // B panel resident in LDS
  __syncthreads();

  v8f acc = {};
  for (int kc = 0; kc < K; kc += 32) {
    // ---- A fragment: two contiguous 8-float runs of one row (4x b128) ----
    float4 a0 = *(const float4*)(Arow + kc + lhi * 8);
    float4 a1 = *(const float4*)(Arow + kc + lhi * 8 + 4);
    float4 a2 = *(const float4*)(Arow + kc + 16 + lhi * 8);
    float4 a3 = *(const float4*)(Arow + kc + 16 + lhi * 8 + 4);
    v16h a;
    a[0]  = (_Float16)a0.x; a[1]  = (_Float16)a0.y;
    a[2]  = (_Float16)a0.z; a[3]  = (_Float16)a0.w;
    a[4]  = (_Float16)a1.x; a[5]  = (_Float16)a1.y;
    a[6]  = (_Float16)a1.z; a[7]  = (_Float16)a1.w;
    a[8]  = (_Float16)a2.x; a[9]  = (_Float16)a2.y;
    a[10] = (_Float16)a2.z; a[11] = (_Float16)a2.w;
    a[12] = (_Float16)a3.x; a[13] = (_Float16)a3.y;
    a[14] = (_Float16)a3.z; a[15] = (_Float16)a3.w;

    // ---- B fragment: lane = col, rows kc+lhi*8 .. +7 and kc+16+lhi*8 .. +7 ----
    v16h b;
    #pragma unroll
    for (int i = 0; i < 16; ++i)
      b[i] = (_Float16)lb[(kc + lhi * 8 + (i & 7) + 16 * (i >> 3)) * 16 + lm];

    acc = __builtin_amdgcn_wmma_f32_16x16x32_f16(
        /*neg_a=*/false, a, /*neg_b=*/false, b,
        /*c_mod=*/(short)0, acc, /*reuse_a=*/false, /*reuse_b=*/false);
  }
  // C/D layout: VGPR j -> M = m0 + j + 8*lhi, N = n0 + lm
  #pragma unroll
  for (int j = 0; j < 8; ++j) {
    int mm = m0 + j + 8 * lhi;
    float v = acc[j] + bv[j];
    if (ACT == 1) {
      // branch-free stable softplus: max(x,0) + log(1 + exp(-|x|))
      v = fmaxf(v, 0.0f) + __logf(1.0f + __expf(-fabsf(v)));
    }
    Out[(size_t)mm * N + (n0 + lm)] = v;
  }
}

// ---------------------- weight padding (run once per call) -----------------
__global__ void k_pad_xw(const float* __restrict__ src,   // [KD_][36][128]
                         float* __restrict__ dst) {       // [KD_][48][128]
  int i = blockIdx.x * blockDim.x + threadIdx.x;
  if (i >= KD_ * XPCP_ * D_) return;
  int k = i / (XPCP_ * D_);
  int rem = i - k * XPCP_ * D_;
  int r = rem / D_, c = rem - r * D_;
  dst[i] = (r < XPC_) ? src[(k * XPC_ + r) * D_ + c] : 0.0f;
}

__global__ void k_pad_dtw(const float* __restrict__ src,  // [KD_][128][4]
                          float* __restrict__ dst) {      // [KD_][128][32]
  int i = blockIdx.x * blockDim.x + threadIdx.x;
  if (i >= KD_ * D_ * KDT_) return;
  int k = i / (D_ * KDT_);
  int rem = i - k * D_ * KDT_;
  int r = rem / KDT_, c = rem - r * KDT_;
  dst[i] = (c < R_) ? src[(k * D_ + r) * R_ + c] : 0.0f;
}

// ------------------------- per-pixel LayerNorm over C ----------------------
__global__ void k_layernorm_c(const float* __restrict__ x,
                              const float* __restrict__ g,
                              const float* __restrict__ b,
                              float* __restrict__ o) {
  int l = blockIdx.x * blockDim.x + threadIdx.x;
  if (l >= L_) return;
  float mu = 0.f;
  for (int c = 0; c < C_; ++c) mu += x[c * L_ + l];
  mu *= (1.0f / C_);
  float var = 0.f;
  for (int c = 0; c < C_; ++c) { float d = x[c * L_ + l] - mu; var += d * d; }
  var *= (1.0f / C_);
  float inv = rsqrtf(var + 1e-6f);
  for (int c = 0; c < C_; ++c)
    o[c * L_ + l] = (x[c * L_ + l] - mu) * inv * g[c] + b[c];
}

// ------------------------- per-pixel KL over channels ----------------------
__global__ void k_kl(const float* __restrict__ p, const float* __restrict__ q,
                     float* __restrict__ out) {
  int l = blockIdx.x * blockDim.x + threadIdx.x;
  if (l >= L_) return;
  float mp = -1e30f, mq = -1e30f;
  for (int c = 0; c < C_; ++c) {
    mp = fmaxf(mp, p[c * L_ + l]);
    mq = fmaxf(mq, q[c * L_ + l]);
  }
  float sp = 0.f, sq = 0.f;
  for (int c = 0; c < C_; ++c) {
    sp += __expf(p[c * L_ + l] - mp);
    sq += __expf(q[c * L_ + l] - mq);
  }
  float lsp = mp + __logf(sp), lsq = mq + __logf(sq);
  for (int c = 0; c < C_; ++c) {
    float pv = p[c * L_ + l];
    out[c * L_ + l] = __expf(pv - lsp) * ((pv - lsp) - (q[c * L_ + l] - lsq));
  }
}

// --------------------- depthwise 3x3 SAME conv + bias + SiLU ---------------
__global__ void k_dwconv_silu(const float* __restrict__ xin,  // [D_][L_]
                              const float* __restrict__ w,    // [D_][9]
                              const float* __restrict__ bias,
                              float* __restrict__ out) {
  int idx = blockIdx.x * blockDim.x + threadIdx.x;
  if (idx >= D_ * L_) return;
  int ch = idx / L_, l = idx - ch * L_;
  int hy = l / W_, wx = l - hy * W_;
  float s = bias[ch];
  #pragma unroll
  for (int dy = -1; dy <= 1; ++dy)
    #pragma unroll
    for (int dx = -1; dx <= 1; ++dx) {
      int yy = hy + dy, xx = wx + dx;
      if (yy >= 0 && yy < H_ && xx >= 0 && xx < W_)
        s += w[ch * 9 + (dy + 1) * 3 + (dx + 1)] * xin[ch * L_ + yy * W_ + xx];
    }
  out[idx] = s * (1.0f / (1.0f + __expf(-s)));
}

// ------------------- build 4 direction orderings xs[k][d][l] ---------------
__global__ void k_build_xs(const float* __restrict__ xc, float* __restrict__ xs) {
  int idx = blockIdx.x * blockDim.x + threadIdx.x;
  if (idx >= D_ * L_) return;
  int ch = idx / L_, l = idx - ch * L_;
  int lr = L_ - 1 - l;
  int lt  = (l  % H_) * W_ + (l  / H_);   // transpose flatten
  int ltr = (lr % H_) * W_ + (lr / H_);
  const float* xr = xc + ch * L_;
  xs[(0 * D_ + ch) * L_ + l] = xr[l];
  xs[(1 * D_ + ch) * L_ + l] = xr[lt];
  xs[(2 * D_ + ch) * L_ + l] = xr[lr];
  xs[(3 * D_ + ch) * L_ + l] = xr[ltr];
}

// ------------------------- selective scan (16 states) ----------------------
__global__ void k_scan(const float* __restrict__ delta,  // [KD_][D_][L_]
                       const float* __restrict__ xdbl,   // [KD_][XPCP_][L_]
                       const float* __restrict__ xs,     // [KD_][D_][L_]
                       const float* __restrict__ A_log,  // [KD_][D_][NS_]
                       const float* __restrict__ Dp,     // [KD_][D_]
                       float* __restrict__ y) {
  int idx = blockIdx.x * blockDim.x + threadIdx.x;
  if (idx >= KD_ * D_) return;
  int k = idx / D_, ch = idx - k * D_;
  float Ac[NS_];
  #pragma unroll
  for (int n = 0; n < NS_; ++n) Ac[n] = -__expf(A_log[(k * D_ + ch) * NS_ + n]);
  float Dv = Dp[k * D_ + ch];
  const float* dl = delta + (size_t)(k * D_ + ch) * L_;
  const float* xv = xs + (size_t)(k * D_ + ch) * L_;
  const float* Bm = xdbl + (size_t)(k * XPCP_ + R_) * L_;
  const float* Cm = xdbl + (size_t)(k * XPCP_ + R_ + NS_) * L_;
  float* yo = y + (size_t)(k * D_ + ch) * L_;
  float h[NS_];
  #pragma unroll
  for (int n = 0; n < NS_; ++n) h[n] = 0.f;
  for (int l = 0; l < L_; ++l) {
    __builtin_prefetch(dl + l + 128, 0, 0);   // global_prefetch_b8
    __builtin_prefetch(xv + l + 128, 0, 0);
    float dt = dl[l], xl = xv[l], accv = 0.f;
    #pragma unroll
    for (int n = 0; n < NS_; ++n) {
      float a = __expf(dt * Ac[n]);
      h[n] = a * h[n] + dt * Bm[n * L_ + l] * xl;
      accv += h[n] * Cm[n * L_ + l];
    }
    yo[l] = accv + Dv * xl;
  }
}

// --------------- merge 4 directions + LayerNorm(d) + SiLU gate -------------
__global__ void k_combine(const float* __restrict__ y,   // [KD_][D_][L_]
                          const float* __restrict__ z,   // [D_][L_]
                          const float* __restrict__ g,
                          const float* __restrict__ b,
                          float* __restrict__ yg) {
  int l = blockIdx.x, dd = threadIdx.x;
  int hy = l / W_, wx = l - hy * W_;
  int lt = wx * H_ + hy;   // untrans pick index
  float v = y[(0 * D_ + dd) * L_ + l]
          + y[(2 * D_ + dd) * L_ + (L_ - 1 - l)]
          + y[(1 * D_ + dd) * L_ + lt]
          + y[(3 * D_ + dd) * L_ + (L_ - 1 - lt)];
  __shared__ float red[D_];
  red[dd] = v;
  __syncthreads();
  for (int s = D_ / 2; s > 0; s >>= 1) {
    if (dd < s) red[dd] += red[dd + s];
    __syncthreads();
  }
  float mean = red[0] * (1.0f / D_);
  __syncthreads();
  float dv = v - mean;
  red[dd] = dv * dv;
  __syncthreads();
  for (int s = D_ / 2; s > 0; s >>= 1) {
    if (dd < s) red[dd] += red[dd + s];
    __syncthreads();
  }
  float var = red[0] * (1.0f / D_);
  float yn = dv * rsqrtf(var + 1e-6f) * g[dd] + b[dd];
  float zl = z[dd * L_ + l];
  yg[dd * L_ + l] = yn * (zl * (1.0f / (1.0f + __expf(-zl))));
}

// --------------------------- gated state update ----------------------------
__global__ void k_update(const float* __restrict__ cur,
                         const float* __restrict__ sig,
                         const float* __restrict__ bia,
                         float* __restrict__ nxt, int n) {
  int i = blockIdx.x * blockDim.x + threadIdx.x;
  if (i < n) nxt[i] = cur[i] * __expf(sig[i]) + bia[i];
}

__global__ void k_sin(const float* __restrict__ x, float* __restrict__ o, int n) {
  int i = blockIdx.x * blockDim.x + threadIdx.x;
  if (i < n) o[i] = sinf(x[i]);
}

// ---------------------------------------------------------------------------
struct MambaParams {
  const float *in_proj_w, *in_proj_b, *conv_w, *conv_b, *x_proj_w,
              *dt_w, *dt_b, *A_log, *D, *out_norm_g, *out_norm_b,
              *out_proj_w, *out_proj_b;
};

static MambaParams load_params(void* const* d_in, int base) {
  MambaParams p;
  p.in_proj_w  = (const float*)d_in[base + 0];
  p.in_proj_b  = (const float*)d_in[base + 1];
  p.conv_w     = (const float*)d_in[base + 2];
  p.conv_b     = (const float*)d_in[base + 3];
  p.x_proj_w   = (const float*)d_in[base + 4];
  p.dt_w       = (const float*)d_in[base + 5];
  p.dt_b       = (const float*)d_in[base + 6];
  p.A_log      = (const float*)d_in[base + 7];
  p.D          = (const float*)d_in[base + 8];
  p.out_norm_g = (const float*)d_in[base + 9];
  p.out_norm_b = (const float*)d_in[base + 10];
  p.out_proj_w = (const float*)d_in[base + 11];
  p.out_proj_b = (const float*)d_in[base + 12];
  return p;
}

struct Ws {
  float *xz, *xc, *xs, *xdbl, *delta, *ybuf, *yg, *xwp, *dtwp, *zbias;
};

static void run_ss2d(const float* x, const MambaParams& p, float* out,
                     const Ws& w, hipStream_t stream) {
  // padded weights for the non-16/32-multiple GEMMs
  k_pad_xw<<<(KD_ * XPCP_ * D_ + 255) / 256, 256, 0, stream>>>(p.x_proj_w, w.xwp);
  k_pad_dtw<<<(KD_ * D_ * KDT_ + 255) / 256, 256, 0, stream>>>(p.dt_w, w.dtwp);

  // in_proj: [256 x 64] * [64 x L] -> xz
  k_wmma_gemm<0><<<dim3(L_ / 16, (2 * D_) / 16), 32, 0, stream>>>(
      p.in_proj_w, x, p.in_proj_b, w.xz, 2 * D_, L_, C_);
  k_dwconv_silu<<<(D_ * L_ + 255) / 256, 256, 0, stream>>>(
      w.xz, p.conv_w, p.conv_b, w.xc);
  k_build_xs<<<(D_ * L_ + 255) / 256, 256, 0, stream>>>(w.xc, w.xs);
  // x_proj per direction: [48 x 128] * [128 x L] -> xdbl (48-row blocks)
  for (int k = 0; k < KD_; ++k)
    k_wmma_gemm<0><<<dim3(L_ / 16, XPCP_ / 16), 32, 0, stream>>>(
        w.xwp + k * XPCP_ * D_, w.xs + (size_t)k * D_ * L_, w.zbias,
        w.xdbl + (size_t)k * XPCP_ * L_, XPCP_, L_, D_);
  // delta per direction: softplus([128 x 32pad] * [32 x L] + dt_b)
  // (pad cols 4..31 of A are zero, so rows 4..31 of B contribute nothing)
  for (int k = 0; k < KD_; ++k)
    k_wmma_gemm<1><<<dim3(L_ / 16, D_ / 16), 32, 0, stream>>>(
        w.dtwp + k * D_ * KDT_, w.xdbl + (size_t)k * XPCP_ * L_,
        p.dt_b + k * D_, w.delta + (size_t)k * D_ * L_, D_, L_, KDT_);
  k_scan<<<(KD_ * D_ + 255) / 256, 256, 0, stream>>>(
      w.delta, w.xdbl, w.xs, p.A_log, p.D, w.ybuf);
  k_combine<<<L_, D_, 0, stream>>>(w.ybuf, w.xz + (size_t)D_ * L_,
                                   p.out_norm_g, p.out_norm_b, w.yg);
  // out_proj: [64 x 128] * [128 x L]
  k_wmma_gemm<0><<<dim3(L_ / 16, C_ / 16), 32, 0, stream>>>(
      p.out_proj_w, w.yg, p.out_proj_b, out, C_, L_, D_);
}

extern "C" void kernel_launch(void* const* d_in, const int* in_sizes, int n_in,
                              void* d_out, int out_size, void* d_ws, size_t ws_size,
                              hipStream_t stream) {
  (void)in_sizes; (void)n_in; (void)out_size; (void)ws_size;
  const float* image = (const float*)d_in[0];
  const float* ng    = (const float*)d_in[1];
  const float* nb    = (const float*)d_in[2];
  MambaParams ps = load_params(d_in, 3);    // params_sigma
  MambaParams pb = load_params(d_in, 16);   // params_bias

  // ---- workspace carve (floats); total ~10.84M floats (~43.3 MB) ----
  float* base = (float*)d_ws;
  size_t off = 0;
  auto take = [&](size_t n) { float* p = base + off; off += n; return p; };
  const size_t CL = (size_t)C_ * L_;
  const size_t DL = (size_t)D_ * L_;
  float* cur0 = take(CL);
  float* cur1 = take(CL);
  float* cur2 = take(CL);   // doubles as zero "previous" for iter 1
  float* kld  = take(CL);
  float* sig  = take(CL);
  float* bia  = take(CL);
  Ws w;
  w.xz    = take(2 * DL);
  w.xc    = take(DL);
  w.xs    = take(4 * DL);
  w.xdbl  = take((size_t)KD_ * XPCP_ * L_);
  w.delta = take(4 * DL);
  w.ybuf  = take(4 * DL);
  w.yg    = take(DL);
  w.xwp   = take((size_t)KD_ * XPCP_ * D_);
  w.dtwp  = take((size_t)KD_ * D_ * KDT_);
  w.zbias = take(256);       // zero bias for the bias-less GEMM

  // current = layernorm2d(image); previous = 0; zero bias vector
  k_layernorm_c<<<(L_ + 255) / 256, 256, 0, stream>>>(image, ng, nb, cur0);
  (void)hipMemsetAsync(cur2, 0, CL * sizeof(float), stream);
  (void)hipMemsetAsync(w.zbias, 0, 256 * sizeof(float), stream);

  // ---- iteration 1: previous = 0 (cur2), current = cur0 ----
  k_kl<<<(L_ + 255) / 256, 256, 0, stream>>>(cur0, cur2, kld);
  run_ss2d(kld,  ps, sig, w, stream);
  run_ss2d(cur0, pb, bia, w, stream);
  k_update<<<(int)((CL + 255) / 256), 256, 0, stream>>>(cur0, sig, bia, cur1, (int)CL);

  // ---- iteration 2: previous = cur0, current = cur1 ----
  k_kl<<<(L_ + 255) / 256, 256, 0, stream>>>(cur1, cur0, kld);
  run_ss2d(kld,  ps, sig, w, stream);
  run_ss2d(cur1, pb, bia, w, stream);
  k_update<<<(int)((CL + 255) / 256), 256, 0, stream>>>(cur1, sig, bia, cur2, (int)CL);

  k_sin<<<(int)((CL + 255) / 256), 256, 0, stream>>>(cur2, (float*)d_out, (int)CL);
}
